// RelativePoseEstimationNetwork_12068858101993
// MI455X (gfx1250) — matrix operand order repaired and checked by reading, not compiled, not run on hardware
//
#include <hip/hip_runtime.h>
#include <math.h>

// ---------- types ----------
typedef __attribute__((ext_vector_type(16))) _Float16 v16h;
typedef __attribute__((ext_vector_type(8)))  float    v8f;
typedef __attribute__((ext_vector_type(4)))  float    f32x4;
typedef __attribute__((ext_vector_type(4)))  int      i32x4;

#define HDIM 64
#define WDIM 64
#define HW   4096
#define BATCH 4
#define NPIX (BATCH*HW)   // 16384

// d_out layout (floats)
#define OFF_COORD 0
#define OFF_UNC   49152
#define OFF_KFC   65536
#define OFF_KFU   114688
#define OFF_POSE  131072

// workspace byte offsets
#define WS_X0   ((size_t)0)
#define WS_WB5  ((size_t)33554432)
#define WS_S5   ((size_t)42991616)
#define WS_WB6  ((size_t)59768832)
#define WS_S6   ((size_t)62128128)
#define WS_WB7  ((size_t)70516736)
#define WS_S7   ((size_t)70582272)

// ---------- gfx1250 async global->LDS path (guarded; fallback = reg staging)
#if defined(__has_builtin)
#if __has_builtin(__builtin_amdgcn_global_load_async_to_lds_b128)
#define USE_ASYNC_LDS 1
#endif
#endif

#ifdef USE_ASYNC_LDS
__device__ __forceinline__ void async_ld_b128(const void* g, void* l) {
  __builtin_amdgcn_global_load_async_to_lds_b128(
      (__attribute__((address_space(1))) i32x4*)g,
      (__attribute__((address_space(3))) i32x4*)l, 0, 0);
}
__device__ __forceinline__ void wait_async0() {
#if __has_builtin(__builtin_amdgcn_s_wait_asynccnt)
  __builtin_amdgcn_s_wait_asynccnt(0);
#else
  asm volatile("s_wait_asynccnt 0x0" ::: "memory");
#endif
}
#endif

// ---------------------------------------------------------------------------
// Pack: NCHW fp32 -> NHWC f16 (32x32 LDS tile transpose, coalesced both sides)
// grid: (HW/32, C/32, B), block 256
// ---------------------------------------------------------------------------
__global__ __launch_bounds__(256)
void chw_to_hwc_f16(const float* __restrict__ in, _Float16* __restrict__ out,
                    int C) {
  __shared__ float tile[32][33];
  const int tx = threadIdx.x & 31;
  const int ty = threadIdx.x >> 5;          // 0..7
  const int pix0 = blockIdx.x * 32;
  const int c0   = blockIdx.y * 32;
  const int b    = blockIdx.z;
  #pragma unroll
  for (int i = 0; i < 32; i += 8)
    tile[ty + i][tx] = in[((size_t)(b * C + c0 + ty + i)) * HW + pix0 + tx];
  __syncthreads();
  #pragma unroll
  for (int i = 0; i < 32; i += 8)
    out[((size_t)(b * HW + pix0 + ty + i)) * C + c0 + tx] =
        (_Float16)tile[tx][ty + i];
}

// ---------------------------------------------------------------------------
// Pack weights: [O][I][kh][kw] fp32 -> [kh*kw][O][I] f16
// ---------------------------------------------------------------------------
__global__ __launch_bounds__(256)
void pack_w_f16(const float* __restrict__ in, _Float16* __restrict__ out,
                int KK, int O, int I) {
  size_t oid = (size_t)blockIdx.x * 256 + threadIdx.x;
  size_t total = (size_t)KK * O * I;
  if (oid >= total) return;
  int t   = (int)(oid / ((size_t)O * I));
  size_t r = oid % ((size_t)O * I);
  int o   = (int)(r / I);
  int ic  = (int)(r % I);
  out[oid] = (_Float16)in[((size_t)o * I + ic) * KK + t];
}

// ---------------------------------------------------------------------------
// Implicit-GEMM conv with WMMA f16->f32, software-pipelined + double-buffered.
//   in : [B*HW][Cin]  f16 (NHWC)
//   wt : [KS*KS][Cout][Cin] f16
//   out: [B*HW][Cout] f16, bias+optional ReLU fused
// Block = 256 threads (8 waves). Block tile: 128 pixels (M) x 64 oc (N).
// Per-wave register tile 32x32 = 4 WMMAs per K-step (K chunk = 32).
// Staging uses GLOBAL_LOAD_ASYNC_TO_LDS_B128 when available (ASYNCcnt path),
// else register-staged global->VGPR->LDS.
// grid: (NPIX/128, Cout/64)
// ---------------------------------------------------------------------------
template <int KS>
__global__ __launch_bounds__(256)
void conv_wmma(const _Float16* __restrict__ in, const _Float16* __restrict__ wt,
               const float* __restrict__ bias, _Float16* __restrict__ out,
               int Cin, int Cout, int relu) {
  constexpr int PAD = KS / 2;
  __shared__ __align__(16) _Float16 lA[2][128 * 32];   // 16 KB
  __shared__ __align__(16) _Float16 lB[2][64 * 32];    //  8 KB

  const int tid  = threadIdx.x;
  const int lane = tid & 31;
  const int wave = tid >> 5;
  const int wm = wave & 3;
  const int wn = wave >> 2;

  const int m0 = blockIdx.x * 128;     // global pixel base (4096 % 128 == 0)
  const int n0 = blockIdx.y * 64;      // out-channel base
  const int bBase = (m0 >> 12) * HW;   // batch pixel base
  const int pix0 = m0 & (HW - 1);

  // staging thread mapping: each thread owns 8 channels of 2 A-rows + 1 B-row
  const int ra = tid >> 2;             // 0..63
  const int ca = (tid & 3) * 8;

  const int cpk = Cin >> 5;            // K chunks per tap
  const int NS  = KS * KS * cpk;       // total K steps

  v8f acc00 = {}, acc01 = {}, acc10 = {}, acc11 = {};

  union Frag { v16h v; f32x4 q[2]; };

  // fragment LDS element offsets (per ISA 16-bit A/B VGPR layout tables)
  const int half = lane >> 4;
  const int l16  = lane & 15;
  const int aOff0 = (wm * 32 + l16) * 32 + half * 8;       // A frag rows 0..15
  const int aOff1 = aOff0 + 16 * 32;                       // A frag rows 16..31
  const int bOff0 = (wn * 32 + l16) * 32 + half * 8;       // B frag cols 0..15
  const int bOff1 = bOff0 + 16 * 32;                       // B frag cols 16..31

  // per-step global source computation
  auto srcAddrA = [&](int s, int chunk, bool& valid) -> const f32x4* {
    const int t  = s / cpk;
    const int c0 = (s - t * cpk) << 5;
    const int dy = t / KS - PAD;
    const int dx = t % KS - PAD;
    const int pp = pix0 + chunk * 64 + ra;
    const int sy = (pp >> 6) + dy;
    const int sx = (pp & 63) + dx;
    valid = ((unsigned)sy < HDIM) && ((unsigned)sx < WDIM);
    return reinterpret_cast<const f32x4*>(
        in + ((size_t)(bBase + sy * WDIM + sx)) * Cin + c0 + ca);
  };
  auto srcAddrB = [&](int s) -> const f32x4* {
    const int t  = s / cpk;
    const int c0 = (s - t * cpk) << 5;
    return reinterpret_cast<const f32x4*>(
        wt + ((size_t)t * Cout + n0 + ra) * Cin + c0 + ca);
  };

#ifdef USE_ASYNC_LDS
  // stage step s into LDS buffer nb via async DMA (zero-fill OOB rows via ds)
  auto stageAsync = [&](int s, int nb) {
    bool v0, v1;
    const f32x4* pa0 = srcAddrA(s, 0, v0);
    const f32x4* pa1 = srcAddrA(s, 1, v1);
    f32x4* la0 = reinterpret_cast<f32x4*>(&lA[nb][ra * 32 + ca]);
    f32x4* la1 = reinterpret_cast<f32x4*>(&lA[nb][(64 + ra) * 32 + ca]);
    const f32x4 z = {0.f, 0.f, 0.f, 0.f};
    if (v0) async_ld_b128(pa0, la0); else *la0 = z;
    if (v1) async_ld_b128(pa1, la1); else *la1 = z;
    async_ld_b128(srcAddrB(s), &lB[nb][ra * 32 + ca]);
  };
  stageAsync(0, 0);
  wait_async0();
#else
  {
    bool v0, v1;
    const f32x4* pa0 = srcAddrA(0, 0, v0);
    const f32x4* pa1 = srcAddrA(0, 1, v1);
    f32x4 z = {0.f, 0.f, 0.f, 0.f};
    *reinterpret_cast<f32x4*>(&lA[0][(ra)*32 + ca])      = v0 ? *pa0 : z;
    *reinterpret_cast<f32x4*>(&lA[0][(64 + ra)*32 + ca]) = v1 ? *pa1 : z;
    *reinterpret_cast<f32x4*>(&lB[0][ra * 32 + ca])      = *srcAddrB(0);
  }
#endif
  __syncthreads();

  for (int s = 0; s < NS; ++s) {
    const int buf = s & 1;
    const bool more = (s + 1 < NS);

#ifdef USE_ASYNC_LDS
    // kick next step's DMA into the alternate buffer; it runs under the WMMAs
    if (more) stageAsync(s + 1, buf ^ 1);
#else
    f32x4 nA0 = {0.f, 0.f, 0.f, 0.f}, nA1 = nA0, nB = nA0;
    if (more) {
      bool v0, v1;
      const f32x4* pa0 = srcAddrA(s + 1, 0, v0);
      const f32x4* pa1 = srcAddrA(s + 1, 1, v1);
      if (v0) nA0 = *pa0;
      if (v1) nA1 = *pa1;
      nB = *srcAddrB(s + 1);
    }
#endif
    // prefetch step s+2 source lines into GL2 (global_prefetch_b8)
    if (s + 2 < NS) {
      bool v0, v1;
      const f32x4* pa0 = srcAddrA(s + 2, 0, v0);
      const f32x4* pa1 = srcAddrA(s + 2, 1, v1);
      if (v0) __builtin_prefetch(pa0, 0, 1);
      if (v1) __builtin_prefetch(pa1, 0, 1);
      __builtin_prefetch(srcAddrB(s + 2), 0, 1);
    }

    // fragment loads from current buffer + 4 WMMAs (32x32 register tile)
    Frag a0, a1, b0, b1;
    a0.q[0] = *reinterpret_cast<const f32x4*>(&lA[buf][aOff0]);
    a0.q[1] = *reinterpret_cast<const f32x4*>(&lA[buf][aOff0 + 16]);
    a1.q[0] = *reinterpret_cast<const f32x4*>(&lA[buf][aOff1]);
    a1.q[1] = *reinterpret_cast<const f32x4*>(&lA[buf][aOff1 + 16]);
    b0.q[0] = *reinterpret_cast<const f32x4*>(&lB[buf][bOff0]);
    b0.q[1] = *reinterpret_cast<const f32x4*>(&lB[buf][bOff0 + 16]);
    b1.q[0] = *reinterpret_cast<const f32x4*>(&lB[buf][bOff1]);
    b1.q[1] = *reinterpret_cast<const f32x4*>(&lB[buf][bOff1 + 16]);

    acc00 = __builtin_amdgcn_wmma_f32_16x16x32_f16(
        false, a0.v, false, b0.v, (short)0, acc00, false, false);
    acc01 = __builtin_amdgcn_wmma_f32_16x16x32_f16(
        false, a0.v, false, b1.v, (short)0, acc01, false, false);
    acc10 = __builtin_amdgcn_wmma_f32_16x16x32_f16(
        false, a1.v, false, b0.v, (short)0, acc10, false, false);
    acc11 = __builtin_amdgcn_wmma_f32_16x16x32_f16(
        false, a1.v, false, b1.v, (short)0, acc11, false, false);

#ifdef USE_ASYNC_LDS
    if (more) wait_async0();    // next-step DMA landed (after the WMMAs)
#else
    if (more) {
      const int nb = buf ^ 1;
      *reinterpret_cast<f32x4*>(&lA[nb][ra * 32 + ca])        = nA0;
      *reinterpret_cast<f32x4*>(&lA[nb][(64 + ra) * 32 + ca]) = nA1;
      *reinterpret_cast<f32x4*>(&lB[nb][ra * 32 + ca])        = nB;
    }
#endif
    __syncthreads();
  }

  // epilogue: C/D layout -> VGPR r, lanes0-15: M=r; lanes16-31: M=8+r; N=l16
  const float bv0 = bias[n0 + wn * 32 + l16];
  const float bv1 = bias[n0 + wn * 32 + l16 + 16];
  #pragma unroll
  for (int r = 0; r < 8; ++r) {
    const int mA = m0 + wm * 32 + half * 8 + r;       // rows 0..15 block
    const int mB = mA + 16;                           // rows 16..31 block
    const int n  = n0 + wn * 32 + l16;
    float v00 = acc00[r] + bv0, v01 = acc01[r] + bv1;
    float v10 = acc10[r] + bv0, v11 = acc11[r] + bv1;
    if (relu) {
      v00 = fmaxf(v00, 0.f); v01 = fmaxf(v01, 0.f);
      v10 = fmaxf(v10, 0.f); v11 = fmaxf(v11, 0.f);
    }
    out[(size_t)mA * Cout + n]      = (_Float16)v00;
    out[(size_t)mA * Cout + n + 16] = (_Float16)v01;
    out[(size_t)mB * Cout + n]      = (_Float16)v10;
    out[(size_t)mB * Cout + n + 16] = (_Float16)v11;
  }
}

// ---------------------------------------------------------------------------
// Head: 1x1 conv 128->4 + Kalman elementwise epilogue. One thread per pixel.
// ---------------------------------------------------------------------------
__global__ __launch_bounds__(256)
void head_kernel(const _Float16* __restrict__ s7, const float* __restrict__ wsp,
                 const float* __restrict__ bsp, const float* __restrict__ lc,
                 const float* __restrict__ lu, float* __restrict__ out) {
  const int p   = blockIdx.x * 256 + threadIdx.x;   // 0..16383
  const int b   = p >> 12;
  const int pix = p & (HW - 1);
  const _Float16* row = s7 + (size_t)p * 128;
  float a0 = bsp[0], a1 = bsp[1], a2 = bsp[2], a3 = bsp[3];
  #pragma unroll 4
  for (int c = 0; c < 128; ++c) {
    const float v = (float)row[c];
    a0 += v * wsp[c];
    a1 += v * wsp[128 + c];
    a2 += v * wsp[256 + c];
    a3 += v * wsp[384 + c];
  }
  const float cm[3] = {a0, a1, a2};
  const float u  = expf(a3);
  const float lus = lu[b * HW + pix];
  const float lv = lus * lus;
  const float mv = u * u;
  const float wK = lv / (lv + mv);
  #pragma unroll
  for (int ch = 0; ch < 3; ++ch) {
    const size_t idx = (size_t)(b * 3 + ch) * HW + pix;
    out[OFF_COORD + idx] = cm[ch];
    out[OFF_KFC + idx]   = (1.f - wK) * lc[idx] + wK * cm[ch];
  }
  out[OFF_UNC + (size_t)b * HW + pix] = u;
  out[OFF_KFU + (size_t)b * HW + pix] = sqrtf((1.f - wK) * lv);
}

// ---------------------------------------------------------------------------
// Pose: one workgroup per batch. 50 GD steps on the reprojection loss with a
// closed-form Rodrigues gradient; wave32 shuffle + LDS reductions.
// ---------------------------------------------------------------------------
__global__ __launch_bounds__(256)
void pose_kernel(const float* __restrict__ kfc, const float* __restrict__ camk,
                 float* __restrict__ poseOut) {
  const int b    = blockIdx.x;
  const int tid  = threadIdx.x;
  const int lane = tid & 31;
  const int wv   = tid >> 5;
  const float* X = kfc + (size_t)b * 3 * HW;
  const float* Y = X + HW;
  const float* Z = Y + HW;
  const float fx = camk[b * 16 + 0], cx = camk[b * 16 + 2];
  const float fy = camk[b * 16 + 5], cy = camk[b * 16 + 6];

  __shared__ float sp[6];
  __shared__ float sred[8][12];
  __shared__ float scnt, sden;

  if (tid < 6) sp[tid] = 0.f;
  // mask count pre-pass
  float cn = 0.f;
  for (int i = tid; i < HW; i += 256) cn += (Z[i] > 0.001f) ? 1.f : 0.f;
  #pragma unroll
  for (int off = 16; off > 0; off >>= 1) cn += __shfl_xor(cn, off, 32);
  if (lane == 0) sred[wv][0] = cn;
  __syncthreads();
  if (tid == 0) {
    float tot = 0.f;
    for (int w = 0; w < 8; ++w) tot += sred[w][0];
    scnt = tot;
    sden = fmaxf(tot, 1.f);
  }
  __syncthreads();
  const float denom = sden;

  float R[9];
  float n2 = 1e-16f, n = 1e-8f, cO = 1.f, sO = 0.f, tt = 0.5f, ssn = 1.f,
        gate = 0.f;

  for (int it = 0; it < 50; ++it) {
    const float th0 = sp[0], th1 = sp[1], th2 = sp[2];
    const float t0 = sp[3], t1 = sp[4], t2 = sp[5];
    const float n2r = th0 * th0 + th1 * th1 + th2 * th2;
    n2 = fmaxf(n2r, 1e-16f);
    gate = (n2r > 1e-16f) ? 1.f : 0.f;
    n = sqrtf(n2);
    cO = cosf(n);
    sO = sinf(n);
    tt = (1.f - cO) / n2;
    ssn = sO / n;
    R[0] = cO + th0 * th0 * tt;  R[1] = th0 * th1 * tt - th2 * ssn; R[2] = th0 * th2 * tt + th1 * ssn;
    R[3] = th1 * th0 * tt + th2 * ssn; R[4] = cO + th1 * th1 * tt;  R[5] = th1 * th2 * tt - th0 * ssn;
    R[6] = th2 * th0 * tt - th1 * ssn; R[7] = th2 * th1 * tt + th0 * ssn; R[8] = cO + th2 * th2 * tt;

    float g[12];
    #pragma unroll
    for (int i = 0; i < 12; ++i) g[i] = 0.f;

    for (int i = tid; i < HW; i += 256) {
      const float px = X[i], py = Y[i], pz = Z[i];
      const float mask = (pz > 0.001f) ? 1.f : 0.f;
      const float qx = R[0] * px + R[1] * py + R[2] * pz + t0;
      const float qy = R[3] * px + R[4] * py + R[5] * pz + t1;
      const float qz = R[6] * px + R[7] * py + R[8] * pz + t2;
      const float z  = mask > 0.f ? qz : 1.f;
      const float iz = 1.f / z;
      const float u = fx * qx * iz + cx;
      const float v = fy * qy * iz + cy;
      const float eu = u - (float)(i & 63);
      const float ev = v - (float)(i >> 6);
      const float sq = eu * eu + ev * ev;
      const float d  = sqrtf(fmaxf(sq, 1e-12f));
      float w = (sq > 1e-12f) ? (mask / d) : 0.f;
      w /= denom;
      const float gu = w * eu, gv = w * ev;
      const float gqx = gu * fx * iz;
      const float gqy = gv * fy * iz;
      const float gqz = mask * (-(gu * fx * qx + gv * fy * qy) * iz * iz);
      g[0] += gqx * px; g[1] += gqx * py; g[2] += gqx * pz;
      g[3] += gqy * px; g[4] += gqy * py; g[5] += gqy * pz;
      g[6] += gqz * px; g[7] += gqz * py; g[8] += gqz * pz;
      g[9] += gqx; g[10] += gqy; g[11] += gqz;
    }
    #pragma unroll
    for (int off = 16; off > 0; off >>= 1)
      #pragma unroll
      for (int i = 0; i < 12; ++i) g[i] += __shfl_xor(g[i], off, 32);
    if (lane == 0)
      #pragma unroll
      for (int i = 0; i < 12; ++i) sred[wv][i] = g[i];
    __syncthreads();
    if (tid == 0) {
      float G[12];
      #pragma unroll
      for (int i = 0; i < 12; ++i) {
        float s = 0.f;
        for (int w = 0; w < 8; ++w) s += sred[w][i];
        G[i] = s;
      }
      const float th[3] = {sp[0], sp[1], sp[2]};
      const float S[9] = {0.f, -th[2], th[1], th[2], 0.f, -th[0], -th[1], th[0], 0.f};
      float gth[3];
      for (int k = 0; k < 3; ++k) {
        const float dn  = gate * th[k] / n;
        const float dc  = -sO * dn;
        const float dsv = cO * dn;
        const float dn2 = gate * 2.f * th[k];
        const float dtt = (sO * dn * n2 - (1.f - cO) * dn2) / (n2 * n2);
        const float dss = (dsv * n - sO * dn) / n2;
        float dS[9] = {0.f, 0.f, 0.f, 0.f, 0.f, 0.f, 0.f, 0.f, 0.f};
        if (k == 0)      { dS[5] = -1.f; dS[7] = 1.f; }
        else if (k == 1) { dS[2] = 1.f;  dS[6] = -1.f; }
        else             { dS[1] = -1.f; dS[3] = 1.f; }
        float acc = 0.f;
        for (int i = 0; i < 3; ++i)
          for (int j = 0; j < 3; ++j) {
            float dR = dtt * th[i] * th[j] + dss * S[i * 3 + j] +
                       ssn * dS[i * 3 + j];
            if (i == j) dR += dc;
            if (i == k) dR += tt * th[j];
            if (j == k) dR += tt * th[i];
            acc += G[i * 3 + j] * dR;
          }
        gth[k] = acc;
      }
      for (int k = 0; k < 3; ++k) sp[k] -= 0.01f * gth[k];
      for (int k = 0; k < 3; ++k) sp[3 + k] -= 0.01f * G[9 + k];
    }
    __syncthreads();
  }

  if (tid == 0) {
    const float th0 = sp[0], th1 = sp[1], th2 = sp[2];
    const float n2f = fmaxf(th0 * th0 + th1 * th1 + th2 * th2, 1e-16f);
    const float nf = sqrtf(n2f), cf = cosf(nf), sf = sinf(nf);
    const float tf = (1.f - cf) / n2f, sn = sf / nf;
    float T[16] = {1.f, 0.f, 0.f, 0.f, 0.f, 1.f, 0.f, 0.f,
                   0.f, 0.f, 1.f, 0.f, 0.f, 0.f, 0.f, 1.f};
    if (scnt >= 6.f) {
      T[0] = cf + th0 * th0 * tf;  T[1] = th0 * th1 * tf - th2 * sn; T[2] = th0 * th2 * tf + th1 * sn; T[3] = sp[3];
      T[4] = th1 * th0 * tf + th2 * sn; T[5] = cf + th1 * th1 * tf;  T[6] = th1 * th2 * tf - th0 * sn; T[7] = sp[4];
      T[8] = th2 * th0 * tf - th1 * sn; T[9] = th2 * th1 * tf + th0 * sn; T[10] = cf + th2 * th2 * tf; T[11] = sp[5];
    }
    for (int i = 0; i < 16; ++i) poseOut[b * 16 + i] = T[i];
  }
}

// ---------------------------------------------------------------------------
extern "C" void kernel_launch(void* const* d_in, const int* in_sizes, int n_in,
                              void* d_out, int out_size, void* d_ws,
                              size_t ws_size, hipStream_t stream) {
  const float* sfeat = (const float*)d_in[0];
  const float* camk  = (const float*)d_in[3];
  const float* lc    = (const float*)d_in[4];
  const float* lu    = (const float*)d_in[5];
  const float* w5 = (const float*)d_in[6];
  const float* b5 = (const float*)d_in[7];
  const float* w6 = (const float*)d_in[8];
  const float* b6 = (const float*)d_in[9];
  const float* w7 = (const float*)d_in[10];
  const float* b7 = (const float*)d_in[11];
  const float* wsp = (const float*)d_in[12];
  const float* bsp = (const float*)d_in[13];
  float* out = (float*)d_out;

  char* ws = (char*)d_ws;
  _Float16* x0  = (_Float16*)(ws + WS_X0);
  _Float16* wb5 = (_Float16*)(ws + WS_WB5);
  _Float16* s5  = (_Float16*)(ws + WS_S5);
  _Float16* wb6 = (_Float16*)(ws + WS_WB6);
  _Float16* s6  = (_Float16*)(ws + WS_S6);
  _Float16* wb7 = (_Float16*)(ws + WS_WB7);
  _Float16* s7  = (_Float16*)(ws + WS_S7);

  // pack activations + weights to f16 GEMM-friendly layouts
  chw_to_hwc_f16<<<dim3(HW / 32, 1024 / 32, BATCH), 256, 0, stream>>>(sfeat, x0, 1024);
  pack_w_f16<<<(9 * 512 * 1024 + 255) / 256, 256, 0, stream>>>(w5, wb5, 9, 512, 1024);
  pack_w_f16<<<(9 * 256 * 512 + 255) / 256, 256, 0, stream>>>(w6, wb6, 9, 256, 512);
  pack_w_f16<<<(128 * 256 + 255) / 256, 256, 0, stream>>>(w7, wb7, 1, 128, 256);

  // WMMA implicit-GEMM convs (prob_map / flow_uncertainty branches are dead in
  // the reference and skipped entirely)
  conv_wmma<3><<<dim3(NPIX / 128, 512 / 64), 256, 0, stream>>>(x0, wb5, b5, s5, 1024, 512, 1);
  conv_wmma<3><<<dim3(NPIX / 128, 256 / 64), 256, 0, stream>>>(s5, wb6, b6, s6, 512, 256, 1);
  conv_wmma<1><<<dim3(NPIX / 128, 128 / 64), 256, 0, stream>>>(s6, wb7, b7, s7, 256, 128, 1);

  // head conv (128->4) + Kalman epilogue
  head_kernel<<<NPIX / 256, 256, 0, stream>>>(s7, wsp, bsp, lc, lu, out);

  // pose estimation, one workgroup per batch element
  pose_kernel<<<BATCH, 256, 0, stream>>>(out + OFF_KFC, camk, out + OFF_POSE);
}
